// StrucTreeEncoder_1632087572923
// MI455X (gfx1250) — matrix operand order repaired and checked by reading, not compiled, not run on hardware
//
#include <hip/hip_runtime.h>
#include <stdint.h>

// CDNA5 / gfx1250 TreeConv encoder.
// Per wave32: 16 batch rows; all 14 tree_conv passes stay in VGPRs.
// D = A(16x32 f16 weights) x B(32x16 f16 batch columns) + C(f32 bias seed)
// via v_wmma_f32_16x16x32_f16. State is packed f16 (4 dwords/node); the
// cross-half exchange is v_permlanex16 (VALU, no LDS). Weights are scoped
// per phase and the carry keeps a shadow copy, keeping VGPR pressure ~120
// so nothing spills to scratch.

typedef __attribute__((ext_vector_type(16))) _Float16 v16h;
typedef __attribute__((ext_vector_type(2)))  _Float16 v2h;
typedef __attribute__((ext_vector_type(8)))  float    v8f;

#define NLATENT 16
#define NNODE   8

static __device__ __forceinline__ v8f wmma_f16(v16h a, v16h b, v8f c) {
  return __builtin_amdgcn_wmma_f32_16x16x32_f16(
      /*neg_a=*/false, a, /*neg_b=*/false, b,
      /*c_mod=*/(short)0, c, /*reuse_a=*/false, /*reuse_b=*/false);
}

// Cross-half exchange (lane <-> lane^16) as a VALU permute, not LDS.
static __device__ __forceinline__ uint32_t xhalf(uint32_t v) {
#if __has_builtin(__builtin_amdgcn_permlanex16)
  return (uint32_t)__builtin_amdgcn_permlanex16(
      (int)v, (int)v, 0x76543210, 0xfedcba98, /*fi=*/true, /*bc=*/false);
#else
  return (uint32_t)__shfl_xor((int)v, 16, 32);
#endif
}

static __device__ __forceinline__ uint32_t pack2(float a, float b) {
  return __builtin_bit_cast(uint32_t, __builtin_amdgcn_cvt_pkrtz(a, b));
}

static __device__ __forceinline__ v16h frag_from(const uint32_t lo[4],
                                                 const uint32_t hh[4]) {
  union { v16h v; uint32_t u[8]; } f;
#pragma unroll
  for (int t = 0; t < 4; ++t) { f.u[t] = lo[t]; f.u[4 + t] = hh[t]; }
  return f.v;
}

// Rolling carry: p = packed f16 features of this lane's D-half,
// x = partner half-wave's p (shadow).
struct Car { uint32_t p[4]; uint32_t x[4]; };

// Root B fragment: lanes 0-15 supply K0..15 = node features (own + shadow);
// lanes 16-31 carry K16..31 which hit the root A-fragment's zeroed columns,
// so no masking needed. Target shadow recomputed on the fly (4 permlanes).
static __device__ __forceinline__ v16h b_root(const uint32_t p[4]) {
  uint32_t sx[4];
#pragma unroll
  for (int t = 0; t < 4; ++t) sx[t] = xhalf(p[t]);
  return frag_from(p, sx);
}

// cat[target ; carry] B fragment (K0..15 = target, K16..31 = carry):
//   lanes<16 : regs0-3 = own target, regs4-7 = target shadow
//   lanes>=16: regs0-3 = carry shadow, regs4-7 = own carry
static __device__ __forceinline__ v16h b_pair(const uint32_t tp[4],
                                              const Car& cr, bool hi) {
  uint32_t tx[4], lo[4], hh[4];
#pragma unroll
  for (int t = 0; t < 4; ++t) tx[t] = xhalf(tp[t]);
#pragma unroll
  for (int t = 0; t < 4; ++t) {
    lo[t] = hi ? cr.x[t] : tp[t];
    hh[t] = hi ? cr.p[t] : tx[t];
  }
  return frag_from(lo, hh);
}

static __device__ __forceinline__ void sigmoid8(v8f& d) {
#pragma unroll
  for (int j = 0; j < 8; ++j) {
#if __has_builtin(__builtin_amdgcn_tanhf)
    // sigmoid(x) = 0.5 * tanh(x/2) + 0.5 ; v_tanh_f32 is new on gfx1250
    d[j] = __builtin_fmaf(0.5f, __builtin_amdgcn_tanhf(d[j] * 0.5f), 0.5f);
#else
    d[j] = __builtin_amdgcn_rcpf(
        1.0f + __builtin_amdgcn_exp2f(d[j] * -1.44269504088896340736f));
#endif
  }
}

// A-matrix fragment (contiguous k order, matches B fragments):
//   lanes 0-15 : M=lane,    halves 0-15 = K0..15
//   lanes 16-31: M=lane-16, halves 0-15 = K16..31
static __device__ __forceinline__ v16h load_a_frag(const float* __restrict__ W,
                                                   int kdim, int m, bool hi) {
  v16h a;
#pragma unroll
  for (int t = 0; t < 8; ++t) {
    int k0 = (hi ? 16 : 0) + t;
    int k1 = (hi ? 24 : 8) + t;
    a[t]     = (_Float16)(k0 < kdim ? W[m * kdim + k0] : 0.0f);
    a[8 + t] = (_Float16)(k1 < kdim ? W[m * kdim + k1] : 0.0f);
  }
  return a;
}

// One phase = 7 tree_conv passes with one weight set. Weights/biases are
// loaded here so their registers are live only within the phase.
static __device__ __forceinline__ void run_phase(uint32_t s[NNODE][4],
    const float* __restrict__ Wr, const float* __restrict__ br,
    const float* __restrict__ Wd, const float* __restrict__ bd,
    const float* __restrict__ Wu, const float* __restrict__ bu,
    int m, bool hi) {
  v16h Ar = load_a_frag(Wr, 16, m, hi);
  v16h Ad = load_a_frag(Wd, 32, m, hi);
  v16h Au = load_a_frag(Wu, 32, m, hi);
  v8f Cr, Cd, Cu;
#pragma unroll
  for (int j = 0; j < 8; ++j) {
    int f = j + (hi ? 8 : 0);
    Cr[j] = br[f]; Cd[j] = bd[f]; Cu[j] = bu[f];
  }

#pragma unroll 1
  for (int p = 0; p < NNODE - 1; ++p) {
    // root: x = x @ Wr.T + br (no activation)
#pragma unroll
    for (int i = 0; i < NNODE; ++i) {
      v8f d = wmma_f16(Ar, b_root(s[i]), Cr);
#pragma unroll
      for (int t = 0; t < 4; ++t) s[i][t] = pack2(d[2 * t], d[2 * t + 1]);
    }
    // down scan: carry = s[0]; s[i] = sigmoid(Wd @ [s[i]; carry] + bd)
    Car c;
#pragma unroll
    for (int t = 0; t < 4; ++t) { c.p[t] = s[0][t]; c.x[t] = xhalf(c.p[t]); }
#pragma unroll
    for (int i = 1; i < NNODE; ++i) {
      v8f d = wmma_f16(Ad, b_pair(s[i], c, hi), Cd);
      sigmoid8(d);
#pragma unroll
      for (int t = 0; t < 4; ++t) {
        c.p[t] = pack2(d[2 * t], d[2 * t + 1]);
        c.x[t] = xhalf(c.p[t]);
        s[i][t] = c.p[t];
      }
    }
    // up scan: carry already holds s[7] (+shadow) from the down scan.
#pragma unroll
    for (int i = NNODE - 2; i >= 0; --i) {
      v8f d = wmma_f16(Au, b_pair(s[i], c, hi), Cu);
      sigmoid8(d);
#pragma unroll
      for (int t = 0; t < 4; ++t) {
        c.p[t] = pack2(d[2 * t], d[2 * t + 1]);
        c.x[t] = xhalf(c.p[t]);
        s[i][t] = c.p[t];
      }
    }
  }
}

__global__ __launch_bounds__(256) void struc_tree_encoder_kernel(
    const float* __restrict__ x,
    const float* __restrict__ Wsr, const float* __restrict__ bsr,
    const float* __restrict__ Wsd, const float* __restrict__ bsd,
    const float* __restrict__ Wsu, const float* __restrict__ bsu,
    const float* __restrict__ Wcr, const float* __restrict__ bcr,
    const float* __restrict__ Wcd, const float* __restrict__ bcd,
    const float* __restrict__ Wcu, const float* __restrict__ bcu,
    float* __restrict__ out, int nbatch) {
  const int lane = threadIdx.x & 31;
  const int wave = threadIdx.x >> 5;
  const bool hi  = lane >= 16;
  const int  m   = lane & 15;
  const int  row = (blockIdx.x * 8 + wave) * 16 + m;

  // Input tile: lane<16 holds features 0..7 (only feats 0,1 nonzero),
  // lane>=16 holds features 8..15 (all zero).
  uint32_t s[NNODE][4];
#pragma unroll
  for (int i = 0; i < NNODE; ++i)
#pragma unroll
    for (int t = 0; t < 4; ++t) s[i][t] = 0u;
  if (!hi && row < nbatch) {
    const float2* __restrict__ xp = reinterpret_cast<const float2*>(x);
#pragma unroll
    for (int i = 0; i < NNODE; ++i) {
      float2 v = xp[row * NNODE + i];
      s[i][0] = pack2(v.x, v.y);
    }
  }

  run_phase(s, Wsr, bsr, Wsd, bsd, Wsu, bsu, m, hi);  // 7 spread passes
  run_phase(s, Wcr, bcr, Wcd, bcd, Wcu, bcu, m, hi);  // 7 collect passes

  // Emit node 0: lane<16 -> features 0..7 of its row, lane>=16 -> 8..15.
  if (row < nbatch) {
    float o[8];
#pragma unroll
    for (int t = 0; t < 4; ++t) {
      v2h h = __builtin_bit_cast(v2h, s[0][t]);
      o[2 * t]     = (float)h[0];
      o[2 * t + 1] = (float)h[1];
    }
    float4* __restrict__ op =
        reinterpret_cast<float4*>(out + (size_t)row * NLATENT + (hi ? 8 : 0));
    op[0] = make_float4(o[0], o[1], o[2], o[3]);
    op[1] = make_float4(o[4], o[5], o[6], o[7]);
  }
}

extern "C" void kernel_launch(void* const* d_in, const int* in_sizes, int n_in,
                              void* d_out, int out_size, void* d_ws, size_t ws_size,
                              hipStream_t stream) {
  // setup_inputs order:
  // 0:x 1:Ws_root 2:bs_root 3:Ws_down 4:bs_down 5:Ws_up 6:bs_up
  // 7:Wc_root 8:bc_root 9:Wc_down 10:bc_down 11:Wc_up 12:bc_up
  // 13:edge_index (ignored, as in reference) 14:num_node (fixed 8)
  const float* x   = (const float*)d_in[0];
  const float* Wsr = (const float*)d_in[1];
  const float* bsr = (const float*)d_in[2];
  const float* Wsd = (const float*)d_in[3];
  const float* bsd = (const float*)d_in[4];
  const float* Wsu = (const float*)d_in[5];
  const float* bsu = (const float*)d_in[6];
  const float* Wcr = (const float*)d_in[7];
  const float* bcr = (const float*)d_in[8];
  const float* Wcd = (const float*)d_in[9];
  const float* bcd = (const float*)d_in[10];
  const float* Wcu = (const float*)d_in[11];
  const float* bcu = (const float*)d_in[12];
  float* out = (float*)d_out;

  const int nbatch = in_sizes[0] / (NNODE * 2);  // x has B*8*2 elements
  const int rowsPerBlock = 8 * 16;               // 8 waves x 16 rows
  const int grid = (nbatch + rowsPerBlock - 1) / rowsPerBlock;

  struc_tree_encoder_kernel<<<grid, 256, 0, stream>>>(
      x, Wsr, bsr, Wsd, bsd, Wsu, bsu,
      Wcr, bcr, Wcd, bcd, Wcu, bcu, out, nbatch);
}
